// LREN_57921928954560
// MI455X (gfx1250) — matrix operand order repaired and checked by reading, not compiled, not run on hardware
//
#include <hip/hip_runtime.h>

// ---------------- CDNA5 WMMA types ----------------
typedef __attribute__((ext_vector_type(16))) __bf16 v16bf;
typedef __attribute__((ext_vector_type(8)))  float  v8f;

__device__ __forceinline__ v8f wmma_bf16(v16bf a, v16bf b, v8f c) {
  // D = A(16x32 bf16) * B(32x16 bf16) + C(16x16 f32)
  return __builtin_amdgcn_wmma_f32_16x16x32_bf16(
      /*neg_a=*/false, a, /*neg_b=*/false, b,
      /*c_mod=*/(short)0, c, /*reuse_a=*/false, /*reuse_b=*/false);
}

// Async global -> LDS copy, 16 bytes per lane (ASYNCcnt-tracked, GV mode).
__device__ __forceinline__ void async_load_b128(unsigned lds_off, const void* gaddr) {
  asm volatile("global_load_async_to_lds_b128 %0, %1, off"
               :: "v"(lds_off), "v"(gaddr)
               : "memory");
}
__device__ __forceinline__ void wait_async0() {
  asm volatile("s_wait_asynccnt 0x0" ::: "memory");
}

// ---------------- problem constants ----------------
#define BATCH   4096
#define TT      64
#define DIM     16
#define H1      256
#define H2      256
#define LAT     64
#define KD      80            // DIM + LAT
#define ROWS    (BATCH * TT)  // 262144 tokens

// workspace layout (bf16 element offsets)
#define W1P_OFF 0                       // 16 frags  * 512
#define W2P_OFF 8192                    // 128 frags * 512
#define W3P_OFF 73728                   // 32 frags  * 512
#define PF_OFF  90112                   // 64*15 frags * 512
#define PF_STRIDE_T 7680                // 15 frags per t (bf16 elems)

// =====================================================================
// Kernel 1: pack W1/W2/W3 (f32, [N,K] row-major) into bf16 WMMA B-frag
// layout: frag(k_tile,n_tile) -> 32 lanes x 16 elems contiguous per lane.
//   lane: N = n_tile*16 + lane%16 ; elem i: K = k_tile*32 + (lane/16)*8 + i%8 + (i/8)*16
// =====================================================================
__device__ __forceinline__ void pack_one(const float* W, __bf16* dst,
                                         int n_tiles, int kmax, int idx) {
  int frag = idx >> 9;
  int e    = idx & 511;
  int lane = e >> 4;
  int i    = e & 15;
  int k_tile = frag / n_tiles;
  int n_tile = frag - k_tile * n_tiles;
  int n = (n_tile << 4) + (lane & 15);
  int k = (k_tile << 5) + ((lane >> 4) << 3) + (i & 7) + ((i >> 3) << 4);
  float v = (k < kmax) ? W[n * kmax + k] : 0.0f;
  dst[idx] = (__bf16)v;
}

__global__ void __launch_bounds__(256)
pack_weights(const float* __restrict__ W1, const float* __restrict__ W2,
             const float* __restrict__ W3, __bf16* __restrict__ ws) {
  int idx = blockIdx.x * 256 + threadIdx.x;
  if (idx < 8192) {                         // W1: K=16 (padded to 32), N=256
    pack_one(W1, ws + W1P_OFF, 16, 16, idx);
  } else if (idx < 73728) {                 // W2: K=256, N=256
    pack_one(W2, ws + W2P_OFF, 16, 256, idx - 8192);
  } else if (idx < 90112) {                 // W3: K=256, N=64
    pack_one(W3, ws + W3P_OFF, 4, 256, idx - 73728);
  }
}

// =====================================================================
// Kernel 2: P_t = (K^T)^t for t=0..63, f32 chain in LDS (full precision),
// pack each P_t into bf16 B-frags (K dim padded 80->96 : 3 k_tiles,
// N dim 80 : 5 n_tiles).
// =====================================================================
__global__ void __launch_bounds__(512)
kpow_kernel(const float* __restrict__ Km, __bf16* __restrict__ PF) {
  __shared__ float Pa[KD * KD];
  __shared__ float Pb[KD * KD];
  int tid = threadIdx.x;

  for (int e = tid; e < KD * KD; e += 512)
    Pa[e] = ((e / KD) == (e % KD)) ? 1.0f : 0.0f;   // P_0 = I
  __syncthreads();

  for (int t = 0; t < 64; t++) {
    float* cur = (t & 1) ? Pb : Pa;
    float* nxt = (t & 1) ? Pa : Pb;

    // pack cur -> PF[t]
    for (int e = tid; e < 15 * 512; e += 512) {
      int frag = e >> 9;
      int le   = e & 511;
      int lane = le >> 4;
      int i    = le & 15;
      int kt = frag / 5, nt = frag - kt * 5;
      int j = (kt << 5) + ((lane >> 4) << 3) + (i & 7) + ((i >> 3) << 4);   // K index (row of P)
      int l = (nt << 4) + (lane & 15);              // N index (col of P)
      float v = (j < KD) ? cur[j * KD + l] : 0.0f;
      PF[t * PF_STRIDE_T + e] = (__bf16)v;
    }

    if (t < 63) {
      // P_{t+1}[j][l] = sum_m P_t[j][m] * K[l][m]
      for (int e = tid; e < KD * KD; e += 512) {
        int j = e / KD;
        int l = e - j * KD;
        const float* Pr = cur + j * KD;
        const float* Kr = Km + l * KD;
        float s = 0.0f;
        #pragma unroll 8
        for (int m = 0; m < KD; m++) s = fmaf(Pr[m], Kr[m], s);
        nxt[e] = s;
      }
    }
    __syncthreads();
  }
}

// =====================================================================
// Kernel 3: fused MLP. One wave = 16 tokens. 4 waves / block.
//   L1: (16x16 pad32) @ W1^T -> relu -> LDS (bf16 16x256)
//   L2: (16x256)      @ W2^T -> relu -> LDS
//   L3: (16x256)      @ W3^T -> g ; write y = [x | g]
// Fragment layouts per CDNA5 ISA 7.12.2.
// =====================================================================
__global__ void __launch_bounds__(128)
mlp_kernel(const float* __restrict__ X, const float* __restrict__ b1,
           const float* __restrict__ b2,
           const __bf16* __restrict__ W1P, const __bf16* __restrict__ W2P,
           const __bf16* __restrict__ W3P, float* __restrict__ Y) {
  __shared__ __align__(32) __bf16 stage[4][16 * 256];
  int lane = threadIdx.x & 31;
  int wave = threadIdx.x >> 5;
  int row0 = (blockIdx.x * 4 + wave) << 4;   // first token of this wave's tile
  __bf16* st = stage[wave];

  int mrow = lane & 15;   // A-matrix M / D-matrix N
  int half = lane >> 4;   // lane-half selects K sub-range / D M-offset

  // ---------- Layer 1: A from X (K=16 real, 16..31 zero) ----------
  v16bf a1;
  {
    const float* xr = X + (size_t)(row0 + mrow) * DIM;
    int klo = half * 8;
    #pragma unroll
    for (int i = 0; i < 8; i++) a1[i] = (__bf16)xr[klo + i];
    #pragma unroll
    for (int i = 8; i < 16; i++) a1[i] = (__bf16)0.0f;
  }
  #pragma unroll
  for (int nt = 0; nt < 16; nt++) {
    v16bf b = *(const v16bf*)(W1P + (nt << 9) + (lane << 4));
    v8f c = {};
    c = wmma_bf16(a1, b, c);
    float bias = b1[(nt << 4) + mrow];
    #pragma unroll
    for (int r = 0; r < 8; r++) {
      float v = c[r] + bias;
      v = v > 0.0f ? v : 0.0f;
      st[(r + half * 8) * 256 + (nt << 4) + mrow] = (__bf16)v;  // h1
    }
  }

  // ---------- load A2 frags from LDS ----------
  v16bf a2[8];
  #pragma unroll
  for (int kt = 0; kt < 8; kt++) {
    const __bf16* rowp = st + mrow * 256 + (kt << 5);
    int klo = half * 8, khi = 16 + half * 8;
    #pragma unroll
    for (int i = 0; i < 8; i++) a2[kt][i] = rowp[klo + i];
    #pragma unroll
    for (int i = 0; i < 8; i++) a2[kt][8 + i] = rowp[khi + i];
  }

  // ---------- Layer 2 ----------
  #pragma unroll
  for (int nt = 0; nt < 16; nt++) {
    v8f c = {};
    #pragma unroll
    for (int kt = 0; kt < 8; kt++) {
      v16bf b = *(const v16bf*)(W2P + ((((kt << 4) + nt) << 9)) + (lane << 4));
      c = wmma_bf16(a2[kt], b, c);
    }
    float bias = b2[(nt << 4) + mrow];
    #pragma unroll
    for (int r = 0; r < 8; r++) {
      float v = c[r] + bias;
      v = v > 0.0f ? v : 0.0f;
      st[(r + half * 8) * 256 + (nt << 4) + mrow] = (__bf16)v;  // h2
    }
  }

  // ---------- load A3 frags ----------
  v16bf a3[8];
  #pragma unroll
  for (int kt = 0; kt < 8; kt++) {
    const __bf16* rowp = st + mrow * 256 + (kt << 5);
    int klo = half * 8, khi = 16 + half * 8;
    #pragma unroll
    for (int i = 0; i < 8; i++) a3[kt][i] = rowp[klo + i];
    #pragma unroll
    for (int i = 0; i < 8; i++) a3[kt][8 + i] = rowp[khi + i];
  }

  // ---------- Layer 3 + write g into y[:,16:80] ----------
  #pragma unroll
  for (int nt = 0; nt < 4; nt++) {
    v8f c = {};
    #pragma unroll
    for (int kt = 0; kt < 8; kt++) {
      v16bf b = *(const v16bf*)(W3P + ((((kt << 2) + nt) << 9)) + (lane << 4));
      c = wmma_bf16(a3[kt], b, c);
    }
    #pragma unroll
    for (int r = 0; r < 8; r++) {
      int tok = row0 + r + half * 8;
      Y[(size_t)tok * KD + DIM + (nt << 4) + mrow] = c[r];
    }
  }

  // ---------- copy x into y[:,0:16] (2 float4 per lane) ----------
  #pragma unroll
  for (int i = 0; i < 2; i++) {
    int e   = lane * 2 + i;        // 0..63 -> 16 tokens x 4 segments
    int tok = row0 + (e >> 2);
    int seg = (e & 3) << 2;
    float4 v = *(const float4*)(X + (size_t)tok * DIM + seg);
    *(float4*)(Y + (size_t)tok * KD + seg) = v;
  }
}

// =====================================================================
// Kernel 4: y_pred[b,t,:] = y0[b,:] @ P_t   (bf16 WMMA, K padded to 96)
// One wave = 16 batch rows; A-frags loaded once, reused over 64 t.
// P_t fragments (15 KB/step, shared by all 4 waves of the block) are
// staged global->LDS with async b128 copies, double-buffered so the
// t+1 transfer overlaps step t's 15 WMMAs.
// =====================================================================
__global__ void __launch_bounds__(128)
ypred_kernel(const float* __restrict__ Y, const __bf16* __restrict__ PF,
             float* __restrict__ YP) {
  __shared__ __align__(32) __bf16 pbuf[2][PF_STRIDE_T];   // 2 x 15 KB
  int tid  = threadIdx.x;
  int lane = tid & 31;
  int wave = tid >> 5;
  int b0   = (blockIdx.x * 4 + wave) << 4;
  int mrow = lane & 15;
  int half = lane >> 4;

  // A frags from y0 = Y[b, t=0, :]  (K = 0..79, padded to 96)
  v16bf a[3];
  const float* y0 = Y + (size_t)(b0 + mrow) * (TT * KD);
  #pragma unroll
  for (int kt = 0; kt < 3; kt++) {
    int klo = (kt << 5) + half * 8;
    int khi = (kt << 5) + 16 + half * 8;
    #pragma unroll
    for (int i = 0; i < 8; i++) {
      int j = klo + i;
      a[kt][i] = (j < KD) ? (__bf16)y0[j] : (__bf16)0.0f;
    }
    #pragma unroll
    for (int i = 0; i < 8; i++) {
      int j = khi + i;
      a[kt][8 + i] = (j < KD) ? (__bf16)y0[j] : (__bf16)0.0f;
    }
  }

  // ---- stage t=0 fragments into LDS (960 x 16B) ----
  {
    unsigned lb = (unsigned)(uintptr_t)&pbuf[0][0];
    for (int i = tid; i < 960; i += 128)
      async_load_b128(lb + i * 16, PF + i * 8);
    wait_async0();
  }
  __syncthreads();

  for (int t = 0; t < 64; t++) {
    const __bf16* cur = pbuf[t & 1];

    // kick off async staging of t+1 into the other buffer
    if (t < 63) {
      unsigned lb = (unsigned)(uintptr_t)&pbuf[(t + 1) & 1][0];
      const __bf16* src = PF + (size_t)(t + 1) * PF_STRIDE_T;
      for (int i = tid; i < 960; i += 128)
        async_load_b128(lb + i * 16, src + i * 8);
    }

    // compute step t from LDS-resident fragments
    #pragma unroll
    for (int nt = 0; nt < 5; nt++) {
      v8f c = {};
      #pragma unroll
      for (int kt = 0; kt < 3; kt++) {
        v16bf b = *(const v16bf*)(cur + ((kt * 5 + nt) << 9) + (lane << 4));
        c = wmma_bf16(a[kt], b, c);
      }
      int l = (nt << 4) + mrow;
      #pragma unroll
      for (int r = 0; r < 8; r++) {
        int bb = b0 + r + half * 8;
        YP[((size_t)bb * TT + t) * KD + l] = c[r];
      }
    }

    wait_async0();      // t+1 transfer complete (own wave)
    __syncthreads();    // all waves done reading cur / staging visible
  }
}

// =====================================================================
extern "C" void kernel_launch(void* const* d_in, const int* in_sizes, int n_in,
                              void* d_out, int out_size, void* d_ws, size_t ws_size,
                              hipStream_t stream) {
  const float* x  = (const float*)d_in[0];
  const float* W1 = (const float*)d_in[1];
  const float* b1 = (const float*)d_in[2];
  const float* W2 = (const float*)d_in[3];
  const float* b2 = (const float*)d_in[4];
  const float* W3 = (const float*)d_in[5];
  const float* Km = (const float*)d_in[6];
  // d_in[7] = n_shifts (== 64, baked into constants)

  __bf16* ws  = (__bf16*)d_ws;
  __bf16* W1P = ws + W1P_OFF;
  __bf16* W2P = ws + W2P_OFF;
  __bf16* W3P = ws + W3P_OFF;
  __bf16* PF  = ws + PF_OFF;

  float* Y  = (float*)d_out;                  // (B*T, 80)
  float* YP = Y + (size_t)ROWS * KD;          // (B, T, 80)

  pack_weights<<<352, 256, 0, stream>>>(W1, W2, W3, ws);
  kpow_kernel<<<1, 512, 0, stream>>>(Km, PF);
  mlp_kernel<<<ROWS / 64, 128, 0, stream>>>(x, b1, b2, W1P, W2P, W3P, Y);
  ypred_kernel<<<BATCH / 64, 128, 0, stream>>>(Y, PF, YP);
}
